// LinearAttention_Cross_40793599377746
// MI455X (gfx1250) — matrix operand order repaired and checked by reading, not compiled, not run on hardware
//
#include <hip/hip_runtime.h>
#include <cstddef>
#include <cstdint>

typedef __attribute__((ext_vector_type(16))) __bf16 v16bf;
typedef __attribute__((ext_vector_type(8)))  __bf16 v8bf;
typedef __attribute__((ext_vector_type(8)))  float  v8f;

union FragAB { v16bf v; v8bf h[2]; };
union AccU   { v8f v; float f[8]; };

__device__ __forceinline__ __bf16 f2bf(float f) {
  uint32_t u = __builtin_bit_cast(uint32_t, f);
  u += 0x7FFFu + ((u >> 16) & 1u);           // round-to-nearest-even
  uint16_t s = (uint16_t)(u >> 16);
  return __builtin_bit_cast(__bf16, s);
}
__device__ __forceinline__ float bf2f(__bf16 h) {
  uint16_t s = __builtin_bit_cast(uint16_t, h);
  uint32_t u = ((uint32_t)s) << 16;
  return __builtin_bit_cast(float, u);
}

// CDNA5 async copy global->LDS (ASYNCcnt-tracked). Per-lane: 16B each.
__device__ __forceinline__ void async_ld_b128(unsigned lds_off, const void* gaddr) {
  asm volatile("global_load_async_to_lds_b128 %0, %1, off"
               :: "v"(lds_off), "v"(gaddr)
               : "memory");
}
__device__ __forceinline__ void wait_async0() {
  asm volatile("s_wait_asynccnt 0x0" ::: "memory");
}

// ---------------------------------------------------------------- converters
__global__ __launch_bounds__(256) void cvt_flat_kernel(const float* __restrict__ in,
                                                       __bf16* __restrict__ out, int n) {
  int i = blockIdx.x * 256 + threadIdx.x;
  if (i < n) out[i] = f2bf(in[i]);
}

// in: [batch][C][N] f32  ->  out: [batch][N][C] bf16
__global__ __launch_bounds__(256) void transpose_cvt_kernel(const float* __restrict__ in,
                                                            __bf16* __restrict__ out,
                                                            int C, int N) {
  long long base = (long long)blockIdx.z * C * N;
  int idx = blockIdx.x * 256 + threadIdx.x;
  if (idx >= C * N) return;
  int c = idx / N, n = idx - c * N;
  out[base + (long long)n * C + c] = f2bf(in[base + idx]);
}

// ------------------------------------------------------- generic NT bf16 GEMM
// C[M,N] = scale * A[M,K] * B[N,K]^T ; A,B row-major bf16; 2-level batching.
// Each wave computes a 32x32 macro-tile (2x2 WMMA tiles, fragments reused 2x).
// M, N must be multiples of 32; K multiple of 32.
__global__ __launch_bounds__(256) void gemm_nt_kernel(
    const __bf16* __restrict__ A, const __bf16* __restrict__ B, void* __restrict__ Cout,
    int M, int N, int K, int ldc,
    long long sAy, long long sAz, long long sBy, long long sBz,
    long long sCy, long long sCz,
    float scale, int out_bf16, int transposed)
{
  const int wave = threadIdx.x >> 5;
  const int lane = threadIdx.x & 31;
  const int tilesM = M >> 5;
  const int tilesN = N >> 5;
  const int tile = blockIdx.x * 8 + wave;
  if (tile >= tilesM * tilesN) return;          // wave-uniform: EXEC stays full
  const int tm = tile % tilesM;
  const int tn = tile / tilesM;
  const int row0 = tm << 5;
  const int col0 = tn << 5;

  A += (long long)blockIdx.y * sAy + (long long)blockIdx.z * sAz;
  B += (long long)blockIdx.y * sBy + (long long)blockIdx.z * sBz;

  // CDNA5 16-bit A 16x32 layout: lanes 0-15 row r hold K {0..7,16..23},
  // lanes 16-31 hold K {8..15,24..31}.  B 32x16: lane holds one column's
  // K values in the same split (lo lanes K {0..7,16..23}, hi lanes K {8..15,24..31}).
  const int r   = lane & 15;
  const int kbA = (lane & 16) ? 8 : 0;
  const int kbB = (lane & 16) ? 8 : 0;
  const __bf16* pa0 = A + (long long)(row0 + r) * K + kbA;
  const __bf16* pa1 = pa0 + (long long)16 * K;
  const __bf16* pb0 = B + (long long)(col0 + r) * K + kbB;
  const __bf16* pb1 = pb0 + (long long)16 * K;

  AccU acc00, acc01, acc10, acc11;
#pragma unroll
  for (int i = 0; i < 8; ++i) {
    acc00.f[i] = 0.0f; acc01.f[i] = 0.0f; acc10.f[i] = 0.0f; acc11.f[i] = 0.0f;
  }

  for (int k0 = 0; k0 < K; k0 += 32) {
    FragAB a0, a1, b0, b1;
    a0.h[0] = *(const v8bf*)(pa0 + k0);
    a0.h[1] = *(const v8bf*)(pa0 + k0 + 16);
    a1.h[0] = *(const v8bf*)(pa1 + k0);
    a1.h[1] = *(const v8bf*)(pa1 + k0 + 16);
    b0.h[0] = *(const v8bf*)(pb0 + k0);
    b0.h[1] = *(const v8bf*)(pb0 + k0 + 16);
    b1.h[0] = *(const v8bf*)(pb1 + k0);
    b1.h[1] = *(const v8bf*)(pb1 + k0 + 16);
    acc00.v = __builtin_amdgcn_wmma_f32_16x16x32_bf16(false, a0.v, false, b0.v,
                                                      (short)0, acc00.v, false, false);
    acc01.v = __builtin_amdgcn_wmma_f32_16x16x32_bf16(false, a0.v, false, b1.v,
                                                      (short)0, acc01.v, false, false);
    acc10.v = __builtin_amdgcn_wmma_f32_16x16x32_bf16(false, a1.v, false, b0.v,
                                                      (short)0, acc10.v, false, false);
    acc11.v = __builtin_amdgcn_wmma_f32_16x16x32_bf16(false, a1.v, false, b1.v,
                                                      (short)0, acc11.v, false, false);
  }

  const int col = lane & 15;
  const int rb  = (lane & 16) ? 8 : 0;        // D: lane holds col, rows rb..rb+7
  const long long cb = (long long)blockIdx.y * sCy + (long long)blockIdx.z * sCz;

  auto store_tile = [&](const AccU& acc, int ro, int co) {
    if (!transposed) {
      if (out_bf16) {
        __bf16* o = (__bf16*)Cout + cb;
#pragma unroll
        for (int i = 0; i < 8; ++i)
          o[(long long)(ro + rb + i) * ldc + (co + col)] = f2bf(acc.f[i] * scale);
      } else {
        float* o = (float*)Cout + cb;
#pragma unroll
        for (int i = 0; i < 8; ++i)
          o[(long long)(ro + rb + i) * ldc + (co + col)] = acc.f[i] * scale;
      }
    } else {
      if (out_bf16) {
        __bf16* o = (__bf16*)Cout + cb;
#pragma unroll
        for (int i = 0; i < 8; ++i)
          o[(long long)(co + col) * ldc + (ro + rb + i)] = f2bf(acc.f[i] * scale);
      } else {
        float* o = (float*)Cout + cb;
#pragma unroll
        for (int i = 0; i < 8; ++i)
          o[(long long)(co + col) * ldc + (ro + rb + i)] = acc.f[i] * scale;
      }
    }
  };
  store_tile(acc00, row0,      col0);
  store_tile(acc01, row0,      col0 + 16);
  store_tile(acc10, row0 + 16, col0);
  store_tile(acc11, row0 + 16, col0 + 16);
}

// ------------------------------------------- softmax over N per row + transpose
// q: [B*512][N] bf16 (row = b*512 + h*64 + c) -> qT: [b*8+h][N][64] bf16
__global__ __launch_bounds__(256) void softmax_t_kernel(const __bf16* __restrict__ q,
                                                        __bf16* __restrict__ qT, int N) {
  const int row = blockIdx.x;
  const int bh = row >> 6;
  const int c  = row & 63;
  const __bf16* in = q + (long long)row * N;
  __bf16* out = qT + (long long)bh * N * 64 + c;
  __shared__ float red[256];
  const int t = threadIdx.x;

  float m = -3.0e38f;
  for (int n = t; n < N; n += 256) m = fmaxf(m, bf2f(in[n]));
  red[t] = m;
  __syncthreads();
  for (int s = 128; s > 0; s >>= 1) {
    if (t < s) red[t] = fmaxf(red[t], red[t + s]);
    __syncthreads();
  }
  m = red[0];
  __syncthreads();

  float sum = 0.0f;
  for (int n = t; n < N; n += 256) sum += __expf(bf2f(in[n]) - m);
  red[t] = sum;
  __syncthreads();
  for (int s = 128; s > 0; s >>= 1) {
    if (t < s) red[t] += red[t + s];
    __syncthreads();
  }
  const float inv = 1.0f / red[0];

  for (int n = t; n < N; n += 256)
    out[(long long)n * 64] = f2bf(__expf(bf2f(in[n]) - m) * inv);
}

// ---------------------- fused: out = LN(Wo[256,512] @ attnT[n,512]^T + bo) * g
// block: 16 pixels (n0..n0+15), 8 waves x 32 channels = all 256 channels.
// The 16x512 bf16 B tile (16 KB, contiguous) is shared by all 8 waves:
// stage it once into LDS with async global->LDS copies, then ds_load it.
__global__ __launch_bounds__(256) void out_ln_kernel(const __bf16* __restrict__ WoB,
                                                     const __bf16* __restrict__ attnT,
                                                     const float* __restrict__ bo,
                                                     const float* __restrict__ g,
                                                     float* __restrict__ out) {
  const int b    = blockIdx.y;
  const int n0   = blockIdx.x << 4;
  const int wave = threadIdx.x >> 5;
  const int lane = threadIdx.x & 31;
  const int m0   = wave << 5;

  __shared__ __align__(16) __bf16 Bs[16 * 512];   // 16 KB staged B tile
  const char* gsrc = (const char*)(attnT + ((long long)b * 4096 + n0) * 512);
  const unsigned lds0 = (unsigned)(size_t)(&Bs[0]);
#pragma unroll
  for (int j = 0; j < 4; ++j) {
    const int chunk = threadIdx.x + 256 * j;      // 1024 x 16B chunks
    async_ld_b128(lds0 + chunk * 16, gsrc + (size_t)chunk * 16);
  }
  wait_async0();
  __syncthreads();

  const int r   = lane & 15;
  const int kbA = (lane & 16) ? 8 : 0;
  const int kbB = (lane & 16) ? 8 : 0;
  const __bf16* pa0 = WoB + (m0 + r) * 512 + kbA;
  const __bf16* pa1 = pa0 + 16 * 512;
  const __bf16* pb  = &Bs[r * 512 + kbB];

  AccU acc0, acc1;
#pragma unroll
  for (int i = 0; i < 8; ++i) { acc0.f[i] = 0.0f; acc1.f[i] = 0.0f; }

  for (int k0 = 0; k0 < 512; k0 += 32) {
    FragAB a0, a1, bf;
    bf.h[0] = *(const v8bf*)(pb + k0);
    bf.h[1] = *(const v8bf*)(pb + k0 + 16);
    a0.h[0] = *(const v8bf*)(pa0 + k0);
    a0.h[1] = *(const v8bf*)(pa0 + k0 + 16);
    a1.h[0] = *(const v8bf*)(pa1 + k0);
    a1.h[1] = *(const v8bf*)(pa1 + k0 + 16);
    acc0.v = __builtin_amdgcn_wmma_f32_16x16x32_bf16(false, a0.v, false, bf.v,
                                                     (short)0, acc0.v, false, false);
    acc1.v = __builtin_amdgcn_wmma_f32_16x16x32_bf16(false, a1.v, false, bf.v,
                                                     (short)0, acc1.v, false, false);
  }

  __shared__ float ssum[16];
  __shared__ float ssq[16];
  if (threadIdx.x < 16) { ssum[threadIdx.x] = 0.0f; ssq[threadIdx.x] = 0.0f; }
  __syncthreads();

  const int col = lane & 15;
  const int rb  = (lane & 16) ? 8 : 0;
  float v0[8], v1[8];
  float ps = 0.0f, pq = 0.0f;
#pragma unroll
  for (int i = 0; i < 8; ++i) {
    int row = m0 + rb + i;
    v0[i] = acc0.f[i] + bo[row];
    ps += v0[i]; pq += v0[i] * v0[i];
    row = m0 + 16 + rb + i;
    v1[i] = acc1.f[i] + bo[row];
    ps += v1[i]; pq += v1[i] * v1[i];
  }
  atomicAdd(&ssum[col], ps);        // ds_add_f32
  atomicAdd(&ssq[col], pq);
  __syncthreads();

  const float mean = ssum[col] * (1.0f / 256.0f);
  const float var  = ssq[col]  * (1.0f / 256.0f) - mean * mean;
  const float rs   = rsqrtf(var + 1e-5f);
  float* o = out + (long long)b * 256 * 4096 + (n0 + col);
#pragma unroll
  for (int i = 0; i < 8; ++i) {
    int row = m0 + rb + i;
    o[(long long)row * 4096] = (v0[i] - mean) * rs * g[row];
    row = m0 + 16 + rb + i;
    o[(long long)row * 4096] = (v1[i] - mean) * rs * g[row];
  }
}

// ------------------------------------------------------------------- launcher
extern "C" void kernel_launch(void* const* d_in, const int* in_sizes, int n_in,
                              void* d_out, int out_size, void* d_ws, size_t ws_size,
                              hipStream_t stream) {
  const float* x  = (const float*)d_in[0];   // [8,256,64,64]
  const float* ct = (const float*)d_in[1];   // [8,512,32,32]
  const float* Wq = (const float*)d_in[2];   // [512,256]
  const float* Wk = (const float*)d_in[3];   // [512,512]
  const float* Wv = (const float*)d_in[4];   // [512,512]
  const float* Wo = (const float*)d_in[5];   // [256,512]
  const float* bo = (const float*)d_in[6];   // [256]
  const float* g  = (const float*)d_in[7];   // [256]
  float* out = (float*)d_out;
  char* ws = (char*)d_ws;
  (void)in_sizes; (void)n_in; (void)out_size;

  size_t off = 0;
  auto alloc = [&](size_t bytes) { size_t r = off; off += (bytes + 255) & ~(size_t)255; return r; };
  const size_t oWq = alloc(512ull * 256 * 2);
  const size_t oWk = alloc(512ull * 512 * 2);
  const size_t oWv = alloc(512ull * 512 * 2);
  const size_t oWo = alloc(256ull * 512 * 2);
  const size_t oXT = alloc(8ull * 4096 * 256 * 2);   // xT; later reused as k|v
  const size_t oQ  = alloc(8ull * 512 * 4096 * 2);   // q bf16; later reused as attnT
  const size_t oQT = alloc(8ull * 512 * 4096 * 2);   // softmax(q)^T
  const size_t oCT = alloc(8ull * 1024 * 512 * 2);   // content^T
  const size_t oCX = alloc(8ull * 8 * 64 * 64 * 2);  // ctx^T per (b,h)
  const size_t need = off;
  if (ws_size < need) return;                         // cannot run without scratch

  const size_t oK  = oXT;                             // 8.4 MB
  const size_t oV  = oXT + 8ull * 512 * 1024 * 2;     // 8.4 MB (fits in xT's 16.8 MB)
  const size_t oAT = oQ;

  __bf16* WqB = (__bf16*)(ws + oWq);
  __bf16* WkB = (__bf16*)(ws + oWk);
  __bf16* WvB = (__bf16*)(ws + oWv);
  __bf16* WoB = (__bf16*)(ws + oWo);
  __bf16* xT  = (__bf16*)(ws + oXT);
  __bf16* qB  = (__bf16*)(ws + oQ);
  __bf16* qT  = (__bf16*)(ws + oQT);
  __bf16* cT  = (__bf16*)(ws + oCT);
  __bf16* cxB = (__bf16*)(ws + oCX);
  __bf16* kB  = (__bf16*)(ws + oK);
  __bf16* vB  = (__bf16*)(ws + oV);
  __bf16* aT  = (__bf16*)(ws + oAT);

  // weights -> bf16
  cvt_flat_kernel<<<dim3(512),  256, 0, stream>>>(Wq, WqB, 512 * 256);
  cvt_flat_kernel<<<dim3(1024), 256, 0, stream>>>(Wk, WkB, 512 * 512);
  cvt_flat_kernel<<<dim3(1024), 256, 0, stream>>>(Wv, WvB, 512 * 512);
  cvt_flat_kernel<<<dim3(512),  256, 0, stream>>>(Wo, WoB, 256 * 512);

  // xT[b][n][c]
  transpose_cvt_kernel<<<dim3(4096, 1, 8), 256, 0, stream>>>(x, xT, 256, 4096);

  // q[b][512][4096] = Wq * x   (bf16 out)  : 16x128 = 2048 macro-tiles
  gemm_nt_kernel<<<dim3(256, 1, 8), 256, 0, stream>>>(WqB, xT, qB,
      512, 4096, 256, 4096,
      0, 0, 0, 4096ll * 256, 0, 512ll * 4096, 1.0f, 1, 0);

  // softmax over the 4096 spatial positions; write qT[bh][n][c]
  softmax_t_kernel<<<dim3(8 * 512), 256, 0, stream>>>(qB, qT, 4096);

  // contentT[b][m][cd]
  transpose_cvt_kernel<<<dim3(2048, 1, 8), 256, 0, stream>>>(ct, cT, 512, 1024);

  // k,v [b][512][1024] : 16x32 = 512 macro-tiles
  gemm_nt_kernel<<<dim3(64, 1, 8), 256, 0, stream>>>(WkB, cT, kB,
      512, 1024, 512, 1024,
      0, 0, 0, 1024ll * 512, 0, 512ll * 1024, 1.0f, 1, 0);
  gemm_nt_kernel<<<dim3(64, 1, 8), 256, 0, stream>>>(WvB, cT, vB,
      512, 1024, 512, 1024,
      0, 0, 0, 1024ll * 512, 0, 512ll * 1024, 1.0f, 1, 0);

  // ctxT[bh][d][c] = (1/1024) sum_m v[d,m] k[c,m] : 2x2 = 4 macro-tiles
  gemm_nt_kernel<<<dim3(1, 8, 8), 256, 0, stream>>>(vB, kB, cxB,
      64, 64, 1024, 64,
      64ll * 1024, 512ll * 1024, 64ll * 1024, 512ll * 1024, 64ll * 64, 8ll * 64 * 64,
      1.0f / 1024.0f, 1, 0);

  // attnT[b][n][h*64+d] = sum_c ctxT[d,c] * qT[n,c]  (transposed store)
  // 2x128 = 256 macro-tiles
  gemm_nt_kernel<<<dim3(32, 8, 8), 256, 0, stream>>>(cxB, qT, aT,
      64, 4096, 64, 512,
      64ll * 64, 8ll * 64 * 64, 4096ll * 64, 8ll * 4096 * 64, 64ll, 4096ll * 512,
      1.0f, 1, 1);

  // fused Wo GEMM + bias + channel LayerNorm -> f32 output
  out_ln_kernel<<<dim3(256, 8), 256, 0, stream>>>(WoB, aT, bo, g, out);
}
